// MambaLayer_77953656422901
// MI455X (gfx1250) — matrix operand order repaired and checked by reading, not compiled, not run on hardware
//
#include <hip/hip_runtime.h>
#include <math.h>

// ---------------------------------------------------------------------------
// MambaLayer for MI455X (gfx1250, wave32, WMMA 16x16x32 f16 -> f32 accum)
// Conv weights double-buffered through LDS (320KB/WGP) to keep the WMMA
// stream off the global-load latency path.
// ---------------------------------------------------------------------------

typedef _Float16 f16;
typedef __attribute__((ext_vector_type(16))) _Float16 v16h;
typedef __attribute__((ext_vector_type(8)))  _Float16 v8h;
typedef __attribute__((ext_vector_type(8)))  float    v8f;

#define CD    96        // model dim
#define LL    16384     // sequence length (64*64*4)
#define NB    2         // batch
#define BLR   32768     // NB*LL rows
#define DIN   192       // d_inner
#define DSTT  16        // d_state
#define EPSF  1e-5f

// ---------------- WMMA fragment loaders (row-major, K contiguous) ----------
__device__ __forceinline__ v16h mk16(v8h a, v8h b) {
  return __builtin_shufflevector(a, b, 0,1,2,3,4,5,6,7,8,9,10,11,12,13,14,15);
}
// A 16x32 f16: lane m (0-15) holds K {0..7,16..23}; lane m+16 holds {8..15,24..31}
__device__ __forceinline__ v16h load_afrag(const f16* __restrict__ row, int k0, int half) {
  const v8h* p0 = reinterpret_cast<const v8h*>(row + k0 + half * 8);
  const v8h* p1 = reinterpret_cast<const v8h*>(row + k0 + half * 8 + 16);
  return mk16(*p0, *p1);
}
// B 32x16 f16 from W(N,K) row-major: lane n holds K {0..7,16..23}, lane n+16 holds {8..15,24..31}
__device__ __forceinline__ v16h load_bfrag(const f16* row, int k0, int half) {
  const v8h* p0 = reinterpret_cast<const v8h*>(row + k0 + half * 8);
  const v8h* p1 = reinterpret_cast<const v8h*>(row + k0 + half * 8 + 16);
  return mk16(*p0, *p1);
}

// ---------------- generic wave GEMM: C(M,N) = A(M,K) * W(N,K)^T ------------
// one wave computes a 16 x (16*NT) tile; K multiple of 32; N multiple of 16*NT
template <int NT>
__global__ __launch_bounds__(256) void gemm_f16(
    const f16* __restrict__ A, int lda, const f16* __restrict__ W, int ldw,
    float* __restrict__ C, int ldc, int ngrp, int nwaves, int K) {
  int wid = (blockIdx.x * blockDim.x + threadIdx.x) >> 5;
  if (wid >= nwaves) return;                 // wave-uniform exit, EXEC stays full
  int lane = threadIdx.x & 31;
  int half = lane >> 4, lr = lane & 15;
  int mtile = wid / ngrp, ng = wid % ngrp;
  const f16* arow = A + (size_t)(mtile * 16 + lr) * lda;
  v8f zero = {0.f,0.f,0.f,0.f,0.f,0.f,0.f,0.f};
  v8f acc[NT];
#pragma unroll
  for (int j = 0; j < NT; ++j) acc[j] = zero;
  for (int k0 = 0; k0 < K; k0 += 32) {
    v16h af = load_afrag(arow, k0, half);
    v16h bf[NT];
#pragma unroll
    for (int j = 0; j < NT; ++j)
      bf[j] = load_bfrag(W + (size_t)((ng * NT + j) * 16 + lr) * ldw, k0, half);
#pragma unroll
    for (int j = 0; j < NT; ++j)
      acc[j] = __builtin_amdgcn_wmma_f32_16x16x32_f16(
          false, af, false, bf[j], (short)0, acc[j], false, false);
  }
  int rbase = mtile * 16 + half * 8;
#pragma unroll
  for (int j = 0; j < NT; ++j) {
    int col = (ng * NT + j) * 16 + lr;
#pragma unroll
    for (int r = 0; r < 8; ++r)
      C[(size_t)(rbase + r) * ldc + col] = acc[j][r];
  }
}

// ---------------- 3x3x3 conv as tap-decomposed implicit GEMM --------------
// xpad: (B,66,66,6,96) f16 channel-last zero-padded; wt: (27,96,96) f16
// out: planar (B,96,L) f32.
// Tap weights are double-buffered through LDS: the block cooperatively stages
// tap t+1 while all 8 waves consume tap t via ds_load B-fragments.
// Grid is exactly BLR/128 blocks: every wave fully active (EXEC all-ones).
__global__ __launch_bounds__(256) void conv3_gemm(
    const f16* __restrict__ xpad, const f16* __restrict__ wt,
    float* __restrict__ outp) {
  __shared__ __align__(16) f16 sw[2][CD * CD];
  int wid = blockIdx.x * 8 + (threadIdx.x >> 5);
  int lane = threadIdx.x & 31;
  int half = lane >> 4, lr = lane & 15;
  int m = wid * 16 + lr;
  int b = m >> 14, p = m & (LL - 1);
  int x = p >> 8, y = (p >> 2) & 63, z = p & 3;
  v8f zero = {0.f,0.f,0.f,0.f,0.f,0.f,0.f,0.f};
  v8f acc[6];
#pragma unroll
  for (int j = 0; j < 6; ++j) acc[j] = zero;

  // prefill tap 0 weights into LDS buffer 0 (2304 qwords, 9 per thread)
  {
    const unsigned long long* gsrc = (const unsigned long long*)wt;
    unsigned long long* ldst = (unsigned long long*)sw[0];
    for (int i = threadIdx.x; i < (CD * CD) / 4; i += 256) ldst[i] = gsrc[i];
  }
  __syncthreads();

  for (int t = 0; t < 27; ++t) {
    int cur = t & 1;
    // stage next tap's weights into the other LDS buffer (overlaps compute)
    if (t + 1 < 27) {
      const unsigned long long* gsrc =
          (const unsigned long long*)(wt + (size_t)(t + 1) * CD * CD);
      unsigned long long* ldst = (unsigned long long*)sw[cur ^ 1];
      for (int i = threadIdx.x; i < (CD * CD) / 4; i += 256) ldst[i] = gsrc[i];
    }
    int kx = t / 9, ky = (t / 3) % 3, kz = t % 3;
    const f16* arow =
        xpad + ((size_t)((b * 66 + x + kx) * 66 + (y + ky)) * 6 + (z + kz)) * CD;
    // prefetch next tap's A row into the near cache while this tap computes
    if (t < 26) {
      int tn = t + 1;
      int kx2 = tn / 9, ky2 = (tn / 3) % 3, kz2 = tn % 3;
      __builtin_prefetch(
          xpad + ((size_t)((b * 66 + x + kx2) * 66 + (y + ky2)) * 6 + (z + kz2)) * CD,
          0, 3);
    }
    const f16* wl = sw[cur];
#pragma unroll
    for (int kc = 0; kc < 3; ++kc) {
      v16h af = load_afrag(arow, kc * 32, half);
#pragma unroll
      for (int j = 0; j < 6; ++j) {
        v16h bf = load_bfrag(wl + (size_t)(j * 16 + lr) * CD, kc * 32, half);
        acc[j] = __builtin_amdgcn_wmma_f32_16x16x32_f16(
            false, af, false, bf, (short)0, acc[j], false, false);
      }
    }
    __syncthreads();
  }
#pragma unroll
  for (int j = 0; j < 6; ++j) {
    int col = j * 16 + lr;
#pragma unroll
    for (int r = 0; r < 8; ++r) {
      int row = wid * 16 + half * 8 + r;
      int bb = row >> 14, pp = row & (LL - 1);
      outp[((size_t)bb * CD + col) * LL + pp] = acc[j][r];
    }
  }
}

// ---------------- wave reduction ------------------------------------------
__device__ __forceinline__ float wred(float v) {
#pragma unroll
  for (int o = 16; o >= 1; o >>= 1) v += __shfl_xor(v, o, 32);
  return v;
}

// x(B,96,L) planar -> x_flat rows + pos, LayerNorm -> x_norm (+f16 mirror)
__global__ __launch_bounds__(256) void k_build_norm(
    const float* __restrict__ x, const float* __restrict__ pos,
    const float* __restrict__ nw, const float* __restrict__ nb,
    float* __restrict__ xflat, float* __restrict__ xnorm, f16* __restrict__ xn16) {
  int row = blockIdx.x * 8 + (threadIdx.x >> 5);
  int lane = threadIdx.x & 31;
  int b = row >> 14, l = row & (LL - 1);
  float v[3]; int c[3];
  float s = 0.f, sq = 0.f;
#pragma unroll
  for (int j = 0; j < 3; ++j) {
    c[j] = lane + j * 32;
    v[j] = x[((size_t)b * CD + c[j]) * LL + l] + pos[(size_t)l * CD + c[j]];
    xflat[(size_t)row * CD + c[j]] = v[j];
    s += v[j]; sq += v[j] * v[j];
  }
  s = wred(s); sq = wred(sq);
  float mu = s * (1.f / CD);
  float inv = rsqrtf(sq * (1.f / CD) - mu * mu + EPSF);
#pragma unroll
  for (int j = 0; j < 3; ++j) {
    float yv = (v[j] - mu) * inv * nw[c[j]] + nb[c[j]];
    xnorm[(size_t)row * CD + c[j]] = yv;
    xn16[(size_t)row * CD + c[j]] = (f16)yv;
  }
}

// LayerNorm over rows of 96 (f32 -> f32)
__global__ __launch_bounds__(256) void k_ln_rows(
    const float* __restrict__ src, const float* __restrict__ nw,
    const float* __restrict__ nb, float* __restrict__ dst) {
  int row = blockIdx.x * 8 + (threadIdx.x >> 5);
  int lane = threadIdx.x & 31;
  float v[3]; int c[3];
  float s = 0.f, sq = 0.f;
#pragma unroll
  for (int j = 0; j < 3; ++j) {
    c[j] = lane + j * 32;
    v[j] = src[(size_t)row * CD + c[j]];
    s += v[j]; sq += v[j] * v[j];
  }
  s = wred(s); sq = wred(sq);
  float mu = s * (1.f / CD);
  float inv = rsqrtf(sq * (1.f / CD) - mu * mu + EPSF);
#pragma unroll
  for (int j = 0; j < 3; ++j)
    dst[(size_t)row * CD + c[j]] = (v[j] - mu) * inv * nw[c[j]] + nb[c[j]];
}

// depthwise causal conv1d (taps=4) + bias + SiLU; input = xz cols [0,192)
__global__ __launch_bounds__(256) void k_dwconv(
    const float* __restrict__ xz, const float* __restrict__ w,
    const float* __restrict__ bias, float* __restrict__ xcc, f16* __restrict__ xc16) {
  int i = blockIdx.x * blockDim.x + threadIdx.x;
  if (i >= BLR * DIN) return;
  int d = i % DIN, row = i / DIN;
  int l = row & (LL - 1), b = row >> 14;
  float acc = bias[d];
#pragma unroll
  for (int k = 0; k < 4; ++k) {
    int ll = l + k - 3;
    if (ll >= 0) acc += w[d * 4 + k] * xz[(size_t)(b * LL + ll) * 384 + d];
  }
  float sv = acc / (1.f + expf(-acc));
  xcc[i] = sv;
  xc16[i] = (f16)sv;
}

// delta = softplus(dt @ dt_proj_w^T + dt_proj_b), dt = dbl48 cols [0,6)
__global__ __launch_bounds__(256) void k_delta(
    const float* __restrict__ dbl48, const float* __restrict__ dtw,
    const float* __restrict__ dtb, float* __restrict__ delta) {
  int i = blockIdx.x * blockDim.x + threadIdx.x;
  if (i >= BLR * DIN) return;
  int d = i % DIN; size_t row = i / DIN;
  float a = dtb[d];
#pragma unroll
  for (int r = 0; r < 6; ++r) a += dbl48[row * 48 + r] * dtw[d * 6 + r];
  delta[i] = (a > 20.f) ? a : log1pf(expf(a));
}

// selective scan: 1 lane per (b,d,s); 16-lane shuffle reduce; fused D-skip,
// SiLU(z) gating and f16 conversion. Unrolled x4: operand loads for 4 steps
// are batched ahead of the dependent recurrence (critical-path kernel).
__global__ __launch_bounds__(256) void k_scan(
    const float* __restrict__ delta, const float* __restrict__ xcc,
    const float* __restrict__ dbl48, const float* __restrict__ xz,
    const float* __restrict__ A_log, const float* __restrict__ Dskip,
    f16* __restrict__ yg16) {
  int gid = blockIdx.x * blockDim.x + threadIdx.x;
  if (gid >= NB * DIN * DSTT) return;
  int s = gid & 15;
  int chain = gid >> 4;
  int d = chain % DIN, b = chain / DIN;
  float A = -expf(A_log[d * DSTT + s]);
  float Dv = Dskip[d];
  float h = 0.f;
  for (int l0 = 0; l0 < LL; l0 += 4) {
    float dt[4], xv[4], Bv[4], Cv[4], zv[4];
#pragma unroll
    for (int u = 0; u < 4; ++u) {
      size_t row = (size_t)b * LL + l0 + u;
      dt[u] = delta[row * DIN + d];
      xv[u] = xcc[row * DIN + d];
      Bv[u] = dbl48[row * 48 + 6 + s];
      Cv[u] = dbl48[row * 48 + 22 + s];
      zv[u] = xz[row * 384 + DIN + d];
    }
#pragma unroll
    for (int u = 0; u < 4; ++u) {
      h = expf(dt[u] * A) * h + (dt[u] * xv[u]) * Bv[u];
      float part = h * Cv[u];
      part += __shfl_xor(part, 1, 32);
      part += __shfl_xor(part, 2, 32);
      part += __shfl_xor(part, 4, 32);
      part += __shfl_xor(part, 8, 32);
      if (s == 0) {
        float sig = 1.f / (1.f + expf(-zv[u]));
        size_t row = (size_t)b * LL + l0 + u;
        yg16[row * DIN + d] = (f16)((part + Dv * xv[u]) * (zv[u] * sig));
      }
    }
  }
}

// fc1 epilogue: bias + exact GELU, zero padded columns, emit f16
__global__ __launch_bounds__(256) void k_gelu16(
    const float* __restrict__ h1, const float* __restrict__ b1, f16* __restrict__ o) {
  int i = blockIdx.x * blockDim.x + threadIdx.x;
  if (i >= BLR * 32) return;
  int c = i & 31;
  float g = 0.f;
  if (c < 24) {
    float v = h1[i] + b1[c];
    g = 0.5f * v * (1.f + erff(v * 0.70710678f));
  }
  o[i] = (f16)g;
}

// fc2 epilogue: add bias, emit f16 rows (mamba-2 input)
__global__ __launch_bounds__(256) void k_bias16(
    const float* __restrict__ h2, const float* __restrict__ b2, f16* __restrict__ o) {
  int i = blockIdx.x * blockDim.x + threadIdx.x;
  if (i >= BLR * CD) return;
  int c = i % CD;
  o[i] = (f16)(h2[i] + b2[c]);
}

// out = x_flat + gamma*(x_mamba + x_spatial) -> planar (B,96,L)
__global__ __launch_bounds__(256) void k_combine(
    const float* __restrict__ xflat, const float* __restrict__ xm,
    const float* __restrict__ xs, const float* __restrict__ gamma,
    float* __restrict__ outp) {
  int i = blockIdx.x * blockDim.x + threadIdx.x;
  if (i >= BLR * CD) return;
  int c = i % CD; int row = i / CD;
  int b = row >> 14, p = row & (LL - 1);
  outp[((size_t)b * CD + c) * LL + p] = xflat[i] + gamma[c] * (xm[i] + xs[i]);
}

// planar f32 -> padded channel-last f16 (B,66,66,6,96); mode=1 applies BN+leaky
__global__ __launch_bounds__(256) void k_pad(
    const float* __restrict__ src, f16* __restrict__ dst,
    const float* __restrict__ mu, const float* __restrict__ var,
    const float* __restrict__ bw, const float* __restrict__ bb, int mode) {
  int i = blockIdx.x * blockDim.x + threadIdx.x;
  const int TOT = NB * 66 * 66 * 6 * CD;
  if (i >= TOT) return;
  int c = i % CD; int rest = i / CD;
  int pz = rest % 6; rest /= 6;
  int py = rest % 66; rest /= 66;
  int px = rest % 66; int b = rest / 66;
  int x = px - 1, y = py - 1, z = pz - 1;
  float v = 0.f;
  if (x >= 0 && x < 64 && y >= 0 && y < 64 && z >= 0 && z < 4) {
    v = src[((size_t)b * CD + c) * LL + ((x * 64 + y) * 4 + z)];
    if (mode) {
      v = (v - mu[c]) * rsqrtf(var[c] + EPSF) * bw[c] + bb[c];
      v = v < 0.f ? 0.01f * v : v;
    }
  }
  dst[i] = (f16)v;
}

// per-channel mean/var over (B, spatial)
__global__ __launch_bounds__(256) void k_bnstats(
    const float* __restrict__ src, float* __restrict__ mu, float* __restrict__ var) {
  int c = blockIdx.x;
  int tid = threadIdx.x;
  float s = 0.f, sq = 0.f;
  for (int i = tid; i < NB * LL; i += 256) {
    int b = i >> 14, p = i & (LL - 1);
    float v = src[((size_t)b * CD + c) * LL + p];
    s += v; sq += v * v;
  }
  __shared__ float ss[256], sqs[256];
  ss[tid] = s; sqs[tid] = sq;
  __syncthreads();
  for (int st = 128; st > 0; st >>= 1) {
    if (tid < st) { ss[tid] += ss[tid + st]; sqs[tid] += sqs[tid + st]; }
    __syncthreads();
  }
  if (tid == 0) {
    float m = ss[0] * (1.f / (NB * LL));
    mu[c] = m;
    var[c] = sqs[0] * (1.f / (NB * LL)) - m * m;
  }
}

// attn = leaky(bn2(o2) + out) -> f16 rows (BL,96)
__global__ __launch_bounds__(256) void k_attn(
    const float* __restrict__ o2, const float* __restrict__ outp,
    const float* __restrict__ mu, const float* __restrict__ var,
    const float* __restrict__ bw, const float* __restrict__ bb,
    f16* __restrict__ attn16) {
  int i = blockIdx.x * blockDim.x + threadIdx.x;
  if (i >= BLR * CD) return;
  int c = i % CD; int row = i / CD;
  int b = row >> 14, p = row & (LL - 1);
  size_t pi = ((size_t)b * CD + c) * LL + p;
  float v = (o2[pi] - mu[c]) * rsqrtf(var[c] + EPSF) * bw[c] + bb[c];
  float a = v + outp[pi];
  a = a < 0.f ? 0.01f * a : a;
  attn16[i] = (f16)a;
}

// final = out_planar + c8out(rows) + c8_b -> d_out planar (B,96,64,64,4)
__global__ __launch_bounds__(256) void k_final(
    const float* __restrict__ outp, const float* __restrict__ c8o,
    const float* __restrict__ c8b, float* __restrict__ dout) {
  int i = blockIdx.x * blockDim.x + threadIdx.x;
  if (i >= NB * CD * LL) return;
  int b = i / (CD * LL);
  int rem = i % (CD * LL);
  int c = rem / LL, p = rem % LL;
  size_t row = (size_t)b * LL + p;
  dout[i] = outp[i] + c8o[row * CD + c] + c8b[c];
}

// weight packers -----------------------------------------------------------
__global__ __launch_bounds__(256) void k_pack(
    const float* __restrict__ s, f16* __restrict__ d,
    int rows, int cols, int drows, int dcols) {
  int i = blockIdx.x * blockDim.x + threadIdx.x;
  if (i >= drows * dcols) return;
  int r = i / dcols, c = i % dcols;
  d[i] = (r < rows && c < cols) ? (f16)s[(size_t)r * cols + c] : (f16)0.f;
}
// (96,96,27) -> (27,96,96)
__global__ __launch_bounds__(256) void k_packtaps(
    const float* __restrict__ w, f16* __restrict__ wt) {
  int i = blockIdx.x * blockDim.x + threadIdx.x;
  if (i >= 27 * CD * CD) return;
  int t = i / (CD * CD); int rem = i % (CD * CD);
  int o = rem / CD, c = rem % CD;
  wt[i] = (f16)w[(size_t)o * (CD * 27) + c * 27 + t];
}

// ---------------- host side -----------------------------------------------
static inline void run_gemm(hipStream_t st, const f16* A, int lda, const f16* W,
                            int ldw, float* C, int ldc, int M, int N, int K, int NT) {
  int ngrp = N / (16 * NT);
  int nwaves = (M / 16) * ngrp;
  int blocks = (nwaves * 32 + 255) / 256;
  switch (NT) {
    case 2: gemm_f16<2><<<blocks, 256, 0, st>>>(A, lda, W, ldw, C, ldc, ngrp, nwaves, K); break;
    case 3: gemm_f16<3><<<blocks, 256, 0, st>>>(A, lda, W, ldw, C, ldc, ngrp, nwaves, K); break;
    case 4: gemm_f16<4><<<blocks, 256, 0, st>>>(A, lda, W, ldw, C, ldc, ngrp, nwaves, K); break;
    default: gemm_f16<6><<<blocks, 256, 0, st>>>(A, lda, W, ldw, C, ldc, ngrp, nwaves, K); break;
  }
}

extern "C" void kernel_launch(void* const* d_in, const int* in_sizes, int n_in,
                              void* d_out, int out_size, void* d_ws, size_t ws_size,
                              hipStream_t stream) {
  // inputs (setup_inputs order)
  const float* x         = (const float*)d_in[0];
  const float* pos       = (const float*)d_in[1];
  const float* norm_w    = (const float*)d_in[2];
  const float* norm_b    = (const float*)d_in[3];
  const float* norm1_w   = (const float*)d_in[4];
  const float* norm1_b   = (const float*)d_in[5];
  const float* norm2_w   = (const float*)d_in[6];
  const float* norm2_b   = (const float*)d_in[7];
  const float* gamma     = (const float*)d_in[8];
  const float* in_proj_w = (const float*)d_in[9];
  const float* conv1d_w  = (const float*)d_in[10];
  const float* conv1d_b  = (const float*)d_in[11];
  const float* x_proj_w  = (const float*)d_in[12];
  const float* dt_proj_w = (const float*)d_in[13];
  const float* dt_proj_b = (const float*)d_in[14];
  const float* A_log     = (const float*)d_in[15];
  const float* D_skip    = (const float*)d_in[16];
  const float* out_proj_w= (const float*)d_in[17];
  const float* fc1_w     = (const float*)d_in[18];
  const float* fc1_b     = (const float*)d_in[19];
  const float* fc2_w     = (const float*)d_in[20];
  const float* fc2_b     = (const float*)d_in[21];
  const float* c1_w      = (const float*)d_in[22];
  const float* bn1_w     = (const float*)d_in[23];
  const float* bn1_b     = (const float*)d_in[24];
  const float* c2_w      = (const float*)d_in[25];
  const float* bn2_w     = (const float*)d_in[26];
  const float* bn2_b     = (const float*)d_in[27];
  const float* c8_w      = (const float*)d_in[28];
  const float* c8_b      = (const float*)d_in[29];
  float* dout = (float*)d_out;

  // workspace carving (256B aligned)
  size_t off = 0;
  auto alloc = [&](size_t nbytes) -> char* {
    size_t cur = (off + 255) & ~(size_t)255;
    off = cur + nbytes;
    return (char*)d_ws + cur;
  };
  const size_t R96 = (size_t)BLR * CD;
  float* x_flat   = (float*)alloc(R96 * 4);
  float* x_norm   = (float*)alloc(R96 * 4);
  float* x_mamba  = (float*)alloc(R96 * 4);
  float* x_spat   = (float*)alloc(R96 * 4);
  float* m_out    = (float*)alloc(R96 * 4);           // also reused for c8 GEMM out
  float* h2       = (float*)alloc(R96 * 4);
  float* h1       = (float*)alloc((size_t)BLR * 32 * 4);
  float* out_pl   = (float*)alloc(R96 * 4);
  float* o1       = (float*)alloc(R96 * 4);
  float* o2       = (float*)alloc(R96 * 4);
  float* xz       = (float*)alloc((size_t)BLR * 384 * 4);
  float* xcc      = (float*)alloc((size_t)BLR * DIN * 4);
  float* dbl48    = (float*)alloc((size_t)BLR * 48 * 4);
  float* delta    = (float*)alloc((size_t)BLR * DIN * 4);
  float* stats    = (float*)alloc(4 * CD * 4);        // mu1,var1,mu2,var2
  f16* xn16   = (f16*)alloc(R96 * 2);
  f16* h2_16  = (f16*)alloc(R96 * 2);
  f16* attn16 = (f16*)alloc(R96 * 2);
  f16* xc16   = (f16*)alloc((size_t)BLR * DIN * 2);
  f16* yg16   = (f16*)alloc((size_t)BLR * DIN * 2);
  f16* h1_16  = (f16*)alloc((size_t)BLR * 32 * 2);
  f16* xpad   = (f16*)alloc((size_t)NB * 66 * 66 * 6 * CD * 2);
  f16* w_in16 = (f16*)alloc(384 * 96 * 2);
  f16* w_xp16 = (f16*)alloc(48 * 192 * 2);
  f16* w_op16 = (f16*)alloc(96 * 192 * 2);
  f16* w_f116 = (f16*)alloc(32 * 96 * 2);
  f16* w_f216 = (f16*)alloc(96 * 32 * 2);
  f16* w_c816 = (f16*)alloc(96 * 96 * 2);
  f16* wtap1  = (f16*)alloc(27 * 96 * 96 * 2);
  f16* wtap2  = (f16*)alloc(27 * 96 * 96 * 2);
  (void)ws_size; (void)n_in; (void)in_sizes; (void)out_size;

  auto nblk = [](long long n) { return (int)((n + 255) / 256); };

  // pack weights to f16 (with zero padding where shapes need it)
  k_pack<<<nblk(384 * 96), 256, 0, stream>>>(in_proj_w, w_in16, 384, 96, 384, 96);
  k_pack<<<nblk(48 * 192), 256, 0, stream>>>(x_proj_w, w_xp16, 38, 192, 48, 192);
  k_pack<<<nblk(96 * 192), 256, 0, stream>>>(out_proj_w, w_op16, 96, 192, 96, 192);
  k_pack<<<nblk(32 * 96), 256, 0, stream>>>(fc1_w, w_f116, 24, 96, 32, 96);
  k_pack<<<nblk(96 * 32), 256, 0, stream>>>(fc2_w, w_f216, 96, 24, 96, 32);
  k_pack<<<nblk(96 * 96), 256, 0, stream>>>(c8_w, w_c816, 96, 96, 96, 96);
  k_packtaps<<<nblk(27 * 96 * 96), 256, 0, stream>>>(c1_w, wtap1);
  k_packtaps<<<nblk(27 * 96 * 96), 256, 0, stream>>>(c2_w, wtap2);

  // x_flat + pos, LN -> x_norm (+f16)
  k_build_norm<<<BLR / 8, 256, 0, stream>>>(x, pos, norm_w, norm_b, x_flat, x_norm, xn16);

  auto run_mamba = [&](const f16* in16, const float* lnw, const float* lnb, float* dst) {
    run_gemm(stream, in16, CD, w_in16, CD, xz, 384, BLR, 384, CD, 4);
    k_dwconv<<<nblk((long long)BLR * DIN), 256, 0, stream>>>(xz, conv1d_w, conv1d_b, xcc, xc16);
    run_gemm(stream, xc16, DIN, w_xp16, DIN, dbl48, 48, BLR, 48, DIN, 3);
    k_delta<<<nblk((long long)BLR * DIN), 256, 0, stream>>>(dbl48, dt_proj_w, dt_proj_b, delta);
    k_scan<<<nblk(NB * DIN * DSTT), 256, 0, stream>>>(delta, xcc, dbl48, xz, A_log, D_skip, yg16);
    run_gemm(stream, yg16, DIN, w_op16, DIN, m_out, CD, BLR, CD, DIN, 6);
    k_ln_rows<<<BLR / 8, 256, 0, stream>>>(m_out, lnw, lnb, dst);
  };

  // mamba #1 on x_norm
  run_mamba(xn16, norm1_w, norm1_b, x_mamba);

  // MLP branch: fc1 -> gelu -> fc2 (+bias) -> f16 rows
  run_gemm(stream, xn16, CD, w_f116, CD, h1, 32, BLR, 32, CD, 2);
  k_gelu16<<<nblk((long long)BLR * 32), 256, 0, stream>>>(h1, fc1_b, h1_16);
  run_gemm(stream, h1_16, 32, w_f216, 32, h2, CD, BLR, CD, 32, 6);
  k_bias16<<<nblk((long long)BLR * CD), 256, 0, stream>>>(h2, fc2_b, h2_16);

  // mamba #2 on MLP output
  run_mamba(h2_16, norm2_w, norm2_b, x_spat);

  // combine -> planar out
  k_combine<<<nblk((long long)BLR * CD), 256, 0, stream>>>(x_flat, x_mamba, x_spat, gamma, out_pl);

  // conv chain (grid exactly BLR/128 blocks: 8 fully-active waves per block)
  const int PADTOT = NB * 66 * 66 * 6 * CD;
  k_pad<<<nblk(PADTOT), 256, 0, stream>>>(out_pl, xpad, stats, stats + 96, bn1_w, bn1_b, 0);
  conv3_gemm<<<BLR / 128, 256, 0, stream>>>(xpad, wtap1, o1);
  k_bnstats<<<CD, 256, 0, stream>>>(o1, stats, stats + 96);
  k_pad<<<nblk(PADTOT), 256, 0, stream>>>(o1, xpad, stats, stats + 96, bn1_w, bn1_b, 1);
  conv3_gemm<<<BLR / 128, 256, 0, stream>>>(xpad, wtap2, o2);
  k_bnstats<<<CD, 256, 0, stream>>>(o2, stats + 192, stats + 288);
  k_attn<<<nblk((long long)BLR * CD), 256, 0, stream>>>(o2, out_pl, stats + 192, stats + 288,
                                                        bn2_w, bn2_b, attn16);
  // pointwise c8 GEMM (reuse m_out as output) + final add
  run_gemm(stream, attn16, CD, w_c816, CD, m_out, CD, BLR, CD, CD, 6);
  k_final<<<nblk((long long)NB * CD * LL), 256, 0, stream>>>(out_pl, m_out, c8_b, dout);
}
